// PrototypeNet_9732395892834
// MI455X (gfx1250) — matrix operand order repaired and checked by reading, not compiled, not run on hardware
//
#include <hip/hip_runtime.h>
#include <stdint.h>

#define NROWS 16384
#define NCOLS 1000
#define CPAD  1024   // prototypes padded with zero rows -> no bounds checks in GEMM
#define KDIM  1024
#define BK    32
#define LDSK  40     // LDS row stride in bf16 elems (80B: 16B-multiple, conflict-free per half-wave)
#define NSTAGES (KDIM / BK)

typedef __attribute__((ext_vector_type(16))) __bf16 v16bf;
typedef __attribute__((ext_vector_type(8)))  __bf16 v8bf;
typedef __attribute__((ext_vector_type(4)))  __bf16 v4bf;
typedef __attribute__((ext_vector_type(8)))  float  v8f;

union FragBF { v16bf v; v8bf h[2]; };

// ---------------------------------------------------------------------------
// Pass 1: one wave32 per row. Fuses:
//   - exact f32 sum-of-squares -> sq[row]
//   - f32 -> bf16 hi/lo split (bf16x3 emulation operands) -> hi[row,:], lo[row,:]
// Rows in [nrows_src, nrows_pad) are zero-filled (zero-padding for C edge).
// ---------------------------------------------------------------------------
__global__ __launch_bounds__(256)
void convert_rowsq_kernel(const float* __restrict__ in, __bf16* __restrict__ hi,
                          __bf16* __restrict__ lo, float* __restrict__ sq,
                          int nrows_src, int nrows_pad, int d) {
  int gid  = blockIdx.x * blockDim.x + threadIdx.x;
  int row  = gid >> 5;
  int lane = gid & 31;
  if (row >= nrows_pad) return;
  __bf16* hrow = hi + (size_t)row * d;
  __bf16* lrow = lo + (size_t)row * d;
  int nf4 = d >> 2;

  if (row >= nrows_src) {  // zero-pad rows
    v4bf z = {(__bf16)0.f, (__bf16)0.f, (__bf16)0.f, (__bf16)0.f};
    for (int i = lane; i < nf4; i += 32) {
      *(v4bf*)&hrow[i * 4] = z;
      *(v4bf*)&lrow[i * 4] = z;
    }
    if (lane == 0) sq[row] = 0.f;
    return;
  }

  const float4* rp = (const float4*)(in + (size_t)row * d);
  float s = 0.f;
  for (int i = lane; i < nf4; i += 32) {
    float4 v = rp[i];
    s += v.x * v.x + v.y * v.y + v.z * v.z + v.w * v.w;
    v4bf h, l;
    h.x = (__bf16)v.x; l.x = (__bf16)(v.x - (float)h.x);
    h.y = (__bf16)v.y; l.y = (__bf16)(v.y - (float)h.y);
    h.z = (__bf16)v.z; l.z = (__bf16)(v.z - (float)h.z);
    h.w = (__bf16)v.w; l.w = (__bf16)(v.w - (float)h.w);
    *(v4bf*)&hrow[i * 4] = h;
    *(v4bf*)&lrow[i * 4] = l;
  }
  #pragma unroll
  for (int off = 16; off > 0; off >>= 1)
    s += __shfl_xor(s, off, 32);
  if (lane == 0) sq[row] = s;
}

// ---------------------------------------------------------------------------
// CDNA5 async copy: global -> LDS, 16B per lane, tracked by ASYNCcnt (GVS mode)
// ---------------------------------------------------------------------------
__device__ __forceinline__ void async_b128(uint32_t lds_off, const __bf16* gbase,
                                           uint32_t byte_off) {
  asm volatile("global_load_async_to_lds_b128 %0, %1, %2 offset:0"
               :: "v"(lds_off), "v"(byte_off), "s"(gbase)
               : "memory");
}

__device__ __forceinline__ void wait_async0() {
  asm volatile("s_wait_asynccnt 0x0" ::: "memory");
}

// ---------------------------------------------------------------------------
// Pass 2: 128x128 macro-tile GEMM on bf16 hi/lo (bf16x3, f32 accumulate),
// double-buffered LDS fed by async-to-LDS copies, fused distance epilogue.
// ---------------------------------------------------------------------------
__global__ __launch_bounds__(256)
void protonet_wmma_kernel(const __bf16* __restrict__ Xh, const __bf16* __restrict__ Xl,
                          const __bf16* __restrict__ Ph, const __bf16* __restrict__ Pl,
                          const float* __restrict__ xsq, const float* __restrict__ psq,
                          float* __restrict__ out) {
  // [stage][array: Ah,Al,Bh,Bl][128 rows * LDSK]
  __shared__ __bf16 smem[2][4][128 * LDSK];

  const int tid   = threadIdx.x;          // 0..255 (8 waves)
  const int wave  = tid >> 5;
  const int lane  = tid & 31;
  const int laneh = lane >> 4;
  const int lanel = lane & 15;
  const int rowBase = blockIdx.y * 128;
  const int colBase = blockIdx.x * 128;

  const v8f zero = {0.f, 0.f, 0.f, 0.f, 0.f, 0.f, 0.f, 0.f};
  v8f acc[8];
  #pragma unroll
  for (int t = 0; t < 8; ++t) acc[t] = zero;

  const int koffA   = 8 * laneh;          // A frag: K in [koffA,+8) U [16+koffA,+8)
  const int koffB   = 8 * laneh;          // B frag: K in [koffB,+8) U [16+koffB,+8)
  const int ldsRowA = (wave * 16 + lanel) * LDSK;

  // Per-thread async copy slots: 512 16B-chunks per tile array, 2 per thread.
  int idx0 = tid, idx1 = tid + 256;
  int r0 = idx0 >> 2, c0 = idx0 & 3;      // row in tile, 16B chunk within 32-K slab
  int r1 = idx1 >> 2, c1 = idx1 & 3;
  uint32_t lds0 = (uint32_t)(r0 * LDSK + c0 * 8) * 2u;   // byte offset within one array
  uint32_t lds1 = (uint32_t)(r1 * LDSK + c1 * 8) * 2u;
  uint32_t ga0 = (uint32_t)(((rowBase + r0) * KDIM) + c0 * 8) * 2u;  // + k0*2 later
  uint32_t ga1 = (uint32_t)(((rowBase + r1) * KDIM) + c1 * 8) * 2u;
  uint32_t gb0 = (uint32_t)(((colBase + r0) * KDIM) + c0 * 8) * 2u;
  uint32_t gb1 = (uint32_t)(((colBase + r1) * KDIM) + c1 * 8) * 2u;

  uint32_t base[2][4];
  #pragma unroll
  for (int b = 0; b < 2; ++b)
    #pragma unroll
    for (int a = 0; a < 4; ++a)
      base[b][a] = (uint32_t)(uintptr_t)&smem[b][a][0];

  auto issue_stage = [&](int buf, int k0) {
    uint32_t kb = (uint32_t)k0 * 2u;
    async_b128(base[buf][0] + lds0, Xh, ga0 + kb);
    async_b128(base[buf][0] + lds1, Xh, ga1 + kb);
    async_b128(base[buf][1] + lds0, Xl, ga0 + kb);
    async_b128(base[buf][1] + lds1, Xl, ga1 + kb);
    async_b128(base[buf][2] + lds0, Ph, gb0 + kb);
    async_b128(base[buf][2] + lds1, Ph, gb1 + kb);
    async_b128(base[buf][3] + lds0, Pl, gb0 + kb);
    async_b128(base[buf][3] + lds1, Pl, gb1 + kb);
  };

  issue_stage(0, 0);

  for (int s = 0; s < NSTAGES; ++s) {
    const int buf = s & 1;
    wait_async0();        // my copies for stage s are done
    __syncthreads();      // everyone's copies for stage s are visible
    if (s + 1 < NSTAGES) issue_stage(buf ^ 1, (s + 1) * BK);

    const __bf16* Ah = &smem[buf][0][0];
    const __bf16* Al = &smem[buf][1][0];
    const __bf16* Bh = &smem[buf][2][0];
    const __bf16* Bl = &smem[buf][3][0];

    FragBF ah, al;
    ah.h[0] = *(const v8bf*)&Ah[ldsRowA + koffA];
    ah.h[1] = *(const v8bf*)&Ah[ldsRowA + 16 + koffA];
    al.h[0] = *(const v8bf*)&Al[ldsRowA + koffA];
    al.h[1] = *(const v8bf*)&Al[ldsRowA + 16 + koffA];

    #pragma unroll
    for (int t = 0; t < 8; ++t) {
      const int ldsRowB = (t * 16 + lanel) * LDSK;
      FragBF bh, bl;
      bh.h[0] = *(const v8bf*)&Bh[ldsRowB + koffB];
      bh.h[1] = *(const v8bf*)&Bh[ldsRowB + koffB + 16];
      bl.h[0] = *(const v8bf*)&Bl[ldsRowB + koffB];
      bl.h[1] = *(const v8bf*)&Bl[ldsRowB + koffB + 16];
      // bf16x3 split: hi*hi + hi*lo + lo*hi, f32 accumulate
      acc[t] = __builtin_amdgcn_wmma_f32_16x16x32_bf16(
          false, ah.v, false, bh.v, (short)0, acc[t], false, false);
      acc[t] = __builtin_amdgcn_wmma_f32_16x16x32_bf16(
          false, ah.v, false, bl.v, (short)0, acc[t], false, false);
      acc[t] = __builtin_amdgcn_wmma_f32_16x16x32_bf16(
          false, al.v, false, bh.v, (short)0, acc[t], false, false);
    }
    __syncthreads();      // all reads of this buffer done before it is refilled
  }

  // ---- epilogue: logits = 2*cross - ||x||^2 - ||p||^2 ----
  const int mBase = rowBase + wave * 16 + 8 * laneh;
  float xq[8];
  #pragma unroll
  for (int r = 0; r < 8; ++r) xq[r] = xsq[mBase + r];

  #pragma unroll
  for (int t = 0; t < 8; ++t) {
    int c = colBase + t * 16 + lanel;
    if (c < NCOLS) {
      float pq = psq[c];
      #pragma unroll
      for (int r = 0; r < 8; ++r) {
        out[(size_t)(mBase + r) * NCOLS + c] = 2.f * acc[t][r] - xq[r] - pq;
      }
    }
  }
}

// ---------------------------------------------------------------------------
extern "C" void kernel_launch(void* const* d_in, const int* in_sizes, int n_in,
                              void* d_out, int out_size, void* d_ws, size_t ws_size,
                              hipStream_t stream) {
  const float* X = (const float*)d_in[0];   // [16384, 1024] f32
  const float* P = (const float*)d_in[1];   // [1000, 1024]  f32
  float* out = (float*)d_out;               // logits [16384*1000] ++ prototypes [1000*1024]

  // Workspace layout
  char* ws = (char*)d_ws;
  float*  xsq = (float*)ws;                                   //  16384 f32
  float*  psq = (float*)(ws + (size_t)NROWS * 4);             //   1024 f32
  __bf16* Xh  = (__bf16*)(ws + (size_t)(NROWS + CPAD) * 4);   // 16384x1024 bf16
  __bf16* Xl  = Xh + (size_t)NROWS * KDIM;
  __bf16* Ph  = Xl + (size_t)NROWS * KDIM;                    // 1024x1024 bf16 (zero-padded)
  __bf16* Pl  = Ph + (size_t)CPAD * KDIM;

  // Pass 1: fused bf16 hi/lo split + exact f32 row norms (single read of X, P)
  convert_rowsq_kernel<<<(NROWS * 32) / 256, 256, 0, stream>>>(X, Xh, Xl, xsq,
                                                               NROWS, NROWS, KDIM);
  convert_rowsq_kernel<<<(CPAD * 32) / 256, 256, 0, stream>>>(P, Ph, Pl, psq,
                                                              NCOLS, CPAD, KDIM);

  // Pass 2: WMMA GEMM with async-to-LDS double buffering + fused epilogue
  dim3 grid(CPAD / 128, NROWS / 128);       // (8, 128)
  protonet_wmma_kernel<<<grid, 256, 0, stream>>>(Xh, Xl, Ph, Pl, xsq, psq, out);

  // Second tuple element: prototypes passthrough
  hipMemcpyAsync(out + (size_t)NROWS * NCOLS, P,
                 sizeof(float) * (size_t)NCOLS * KDIM,
                 hipMemcpyDeviceToDevice, stream);
}
